// LinearAutoEncoder_47107201303349
// MI455X (gfx1250) — matrix operand order repaired and checked by reading, not compiled, not run on hardware
//
#include <hip/hip_runtime.h>
#include <hip/hip_bf16.h>

// Problem constants (from reference)
#define NROWS 262144
#define POSD  63
#define LATD  128
#define FEAT  191      // POSD + LATD
#define FPAD  192      // padded K (multiple of 32 for bf16 WMMA)
#define NC    256
#define WROWS (3 * NC) // 768

typedef __attribute__((ext_vector_type(16))) __bf16 v16bf;
typedef __attribute__((ext_vector_type(8)))  float  v8f;

// ---------------------------------------------------------------------------
// Kernel 1: pack concat(W_pos, W_feat) -> bf16 Wcat[768][192], zero-padded.
// 288 KB, lives in d_ws; 32B-aligned rows so B fragments are one v16bf load.
// The zero in column 191 makes the K=191 product zero regardless of A.
// ---------------------------------------------------------------------------
__global__ __launch_bounds__(256) void pack_w_kernel(
    const float* __restrict__ Wpos, const float* __restrict__ Wfeat,
    __bf16* __restrict__ Wcat) {
  int idx = blockIdx.x * 256 + threadIdx.x;
  if (idx >= WROWS * FPAD) return;
  int g = idx / FPAD;
  int f = idx - g * FPAD;
  float v = 0.0f;
  if (f < POSD)       v = Wpos[g * POSD + f];
  else if (f < FEAT)  v = Wfeat[g * LATD + (f - POSD)];
  Wcat[idx] = (__bf16)v;
}

// One K=32 step: build A fragment (contiguous K order, matching B: halves 0..15 = K
// koffA..koffA+15), load three gathered B
// fragments (same contiguous K order: lane's 16 halves = K koffB..koffB+15 of
// its column), and accumulate three WMMAs.
// CLAMP (last chunk only): clamp feature index to 190 instead of branching —
// Wcat column 191 is zero, so the K=191 term contributes 0 either way, and
// the clamped load stays in-bounds and finite (no 0*Inf->NaN risk).
template <bool CLAMP>
__device__ __forceinline__ void do_chunk(int kk, const float* __restrict__ xr,
                                         const __bf16* __restrict__ wr,
                                         int koffA, int koffB,
                                         v8f& c0, v8f& c1, v8f& c2) {
  v16bf a;
#pragma unroll
  for (int e = 0; e < 8; ++e) {
    int f1 = kk + koffA + e;
    int f2 = kk + 8 + koffA + e;
    if (CLAMP) {
      f1 = (f1 < FEAT) ? f1 : (FEAT - 1);
      f2 = (f2 < FEAT) ? f2 : (FEAT - 1);
    }
    a[e]     = (__bf16)xr[f1];
    a[e + 8] = (__bf16)xr[f2];
  }
  const __bf16* w0 = wr + kk + koffB;           // 32B-aligned by construction
  v16bf b0 = *(const v16bf*)(w0);
  v16bf b1 = *(const v16bf*)(w0 + FPAD);
  v16bf b2 = *(const v16bf*)(w0 + 2 * FPAD);
  c0 = __builtin_amdgcn_wmma_f32_16x16x32_bf16(false, a, false, b0, (short)0, c0, false, false);
  c1 = __builtin_amdgcn_wmma_f32_16x16x32_bf16(false, a, false, b1, (short)0, c1, false, false);
  c2 = __builtin_amdgcn_wmma_f32_16x16x32_bf16(false, a, false, b2, (short)0, c2, false, false);
}

// Select element i of a v8f accumulator with a cndmask chain (no scratch).
__device__ __forceinline__ float sel8(v8f c, int i) {
  float r = c[0];
#pragma unroll
  for (int k = 1; k < 8; ++k) r = (i == k) ? c[k] : r;
  return r;
}

// ---------------------------------------------------------------------------
// Kernel 2: one wave per 16 rows. Three WMMA chains (j = 0,1,2) whose B
// columns are the gathered weight rows W[3*cid_i + j]; keep tile diagonals.
// ---------------------------------------------------------------------------
__global__ __launch_bounds__(256) void rgb_gather_wmma_kernel(
    const float* __restrict__ X, const int* __restrict__ cids,
    const __bf16* __restrict__ Wcat, float* __restrict__ out) {
  const int l    = threadIdx.x & 31;
  const int wave = threadIdx.x >> 5;
  const int g    = blockIdx.x * 8 + wave;   // 16-row group id
  const int n    = l & 15;                  // this lane's row (A) / column (B)
  const int row  = g * 16 + n;

  const float* xr = X + (long)row * FEAT;
  const int cid   = cids[row];
  const __bf16* wr = Wcat + (long)(3 * cid) * FPAD;

  const int koffA = (l < 16) ? 0 : 16;
  const int koffB = (l < 16) ? 0 : 16;

  v8f c0 = {}, c1 = {}, c2 = {};
  do_chunk<false>(0,   xr, wr, koffA, koffB, c0, c1, c2);
  do_chunk<false>(32,  xr, wr, koffA, koffB, c0, c1, c2);
  do_chunk<false>(64,  xr, wr, koffA, koffB, c0, c1, c2);
  do_chunk<false>(96,  xr, wr, koffA, koffB, c0, c1, c2);
  do_chunk<false>(128, xr, wr, koffA, koffB, c0, c1, c2);
  do_chunk<true >(160, xr, wr, koffA, koffB, c0, c1, c2);  // K=160..191, padded

  // Diagonal D[n,n]: lanes 0..7 hold rows 0..7 (vgpr = lane), lanes 24..31
  // hold rows 8..15 (vgpr = lane-24). vgpr index == lane & 7 in both cases.
  const bool active = (l < 8) || (l >= 24);
  if (active) {
    const int v    = l & 7;
    const int orow = g * 16 + ((l < 8) ? l : (l - 16));
    const long ob  = (long)orow * 3;
    // Write-once output: non-temporal so it doesn't evict streaming X lines.
    __builtin_nontemporal_store(sel8(c0, v), &out[ob + 0]);
    __builtin_nontemporal_store(sel8(c1, v), &out[ob + 1]);
    __builtin_nontemporal_store(sel8(c2, v), &out[ob + 2]);
  }
}

extern "C" void kernel_launch(void* const* d_in, const int* in_sizes, int n_in,
                              void* d_out, int out_size, void* d_ws, size_t ws_size,
                              hipStream_t stream) {
  const float* X     = (const float*)d_in[0];   // [N, 191] f32
  const int*   cids  = (const int*)d_in[1];     // [N] int
  const float* Wpos  = (const float*)d_in[2];   // [768, 63] f32
  const float* Wfeat = (const float*)d_in[3];   // [768, 128] f32
  float*       out   = (float*)d_out;           // [N, 3] f32
  __bf16*      Wcat  = (__bf16*)d_ws;           // [768, 192] bf16 (288 KB)

  pack_w_kernel<<<(WROWS * FPAD + 255) / 256, 256, 0, stream>>>(Wpos, Wfeat, Wcat);

  // 8 waves/block * 16 rows/wave = 128 rows per block; N/128 = 2048 blocks.
  rgb_gather_wmma_kernel<<<NROWS / 128, 256, 0, stream>>>(X, cids, Wcat, out);
}
